// MultiHeadAttentionLayer_28982439313653
// MI455X (gfx1250) — matrix operand order repaired and checked by reading, not compiled, not run on hardware
//
#include <hip/hip_runtime.h>
#include <hip/hip_bf16.h>

typedef __attribute__((ext_vector_type(16))) _Float16 h16;
typedef __attribute__((ext_vector_type(8)))  _Float16 h8;
typedef __attribute__((ext_vector_type(8)))  float    v8f;

// ---------------------------------------------------------------------------
// D = A(16x32 f16) * B(32x16 f16) + C(16x16 f32)
// ---------------------------------------------------------------------------
__device__ __forceinline__ v8f wmma_f16(h16 a, h16 b, v8f c) {
  return __builtin_amdgcn_wmma_f32_16x16x32_f16(
      /*neg_a=*/false, a, /*neg_b=*/false, b,
      /*c_mod=*/(short)0, c, /*reuse_a=*/false, /*reuse_b=*/false);
}

// Build a 16-half fragment from two contiguous 8-half (16B-aligned) runs.
__device__ __forceinline__ h16 ld_h16(const _Float16* p0, const _Float16* p1) {
  h8 x = *(const h8*)p0;
  h8 y = *(const h8*)p1;
  h16 r;
#pragma unroll
  for (int i = 0; i < 8; ++i) { r[i] = x[i]; r[i + 8] = y[i]; }
  return r;
}

// ---------------------------------------------------------------------------
// Dimensions
// ---------------------------------------------------------------------------
// B=8, L=12, N=384, d=128, atten=256, H=8, hd=32, BL=96, blh = BL*H = 768
#define M_ROWS   36864     // B*L*N
#define NQ       384
#define BLH      768
#define HD       32
#define ATT      256
#define DMODEL   128

// ---------------------------------------------------------------------------
// Kernel 0: convert + transpose weights to f16 so WMMA B-fragments (two
// 8-half K runs per lane) are straight 16B loads.
//   WqT/WkT/WvT : [256][128]  (row = output col, contiguous fan_in)
//   WdT         : [128][256]
// ---------------------------------------------------------------------------
__global__ void prep_weights_kernel(const float* __restrict__ Wq,
                                    const float* __restrict__ Wk,
                                    const float* __restrict__ Wv,
                                    const float* __restrict__ Wd,
                                    _Float16* __restrict__ WqT,
                                    _Float16* __restrict__ WkT,
                                    _Float16* __restrict__ WvT,
                                    _Float16* __restrict__ WdT) {
  int i = blockIdx.x * blockDim.x + threadIdx.x;   // 0 .. 4*32768-1
  if (i < 3 * 32768) {
    const float* W = (i < 32768) ? Wq : (i < 2 * 32768 ? Wk : Wv);
    _Float16*    T = (i < 32768) ? WqT : (i < 2 * 32768 ? WkT : WvT);
    int j = i & 32767;
    int r = j >> 8;          // fan_in  0..127
    int c = j & 255;         // fan_out 0..255
    T[c * 128 + r] = (_Float16)W[r * 256 + c];
  } else {
    int j = i - 3 * 32768;
    int r = j >> 7;          // fan_in  0..255
    int c = j & 127;         // fan_out 0..127
    WdT[c * 256 + r] = (_Float16)Wd[r * 128 + c];
  }
}

// ---------------------------------------------------------------------------
// Kernel 1: projection GEMM  Y = X[36864,128] * W[128,256] + b  (f16 out)
// One wave -> 16 rows x 64 cols (4 column tiles), K = 128 in 4 WMMA steps.
// vmode==0 : store [blh][n][hd]   (layout for Q and K)
// vmode==1 : store [blh][hd][n]   (transposed layout for V)
// grid = 2304 blocks x 128 threads (4 waves: same M tile, 4 N strips)
// ---------------------------------------------------------------------------
__global__ void proj_kernel(const float* __restrict__ X,
                            const _Float16* __restrict__ Wt,   // [256][128] f16
                            const float* __restrict__ bias,    // [256] f32
                            _Float16* __restrict__ out,
                            int vmode) {
  int w     = blockIdx.x * 4 + (threadIdx.x >> 5);
  int lane  = threadIdx.x & 31;
  int lg    = lane >> 4;         // half-wave group
  int ln    = lane & 15;
  int mtile = w >> 2;            // 0..2303
  int nbase = (w & 3) * 64;      // 0,64,128,192
  int rbase = mtile * 16;

  const float* xrow = X + (size_t)(rbase + ln) * 128;   // A row = lane%16
  v8f acc[4] = {};

  for (int k = 0; k < 128; k += 32) {
    // A fragment: rows of X, f32 -> f16
    h16 A;
    const float* p0 = xrow + k + lg * 8;
    const float* p1 = p0 + 16;
#pragma unroll
    for (int i = 0; i < 8; ++i) { A[i] = (_Float16)p0[i]; A[i + 8] = (_Float16)p1[i]; }
#pragma unroll
    for (int t = 0; t < 4; ++t) {
      const _Float16* wr = Wt + (size_t)(nbase + t * 16 + ln) * 128 + k + lg * 8;
      acc[t] = wmma_f16(A, ld_h16(wr, wr + 16), acc[t]);
    }
  }

  int bl  = rbase / NQ;          // tile never crosses a bl boundary (384 % 16 == 0)
  int nq0 = rbase % NQ;
#pragma unroll
  for (int t = 0; t < 4; ++t) {
    int   col = nbase + t * 16 + ln;
    float bv  = bias[col];
    int   h   = col >> 5, hd = col & 31;
#pragma unroll
    for (int r = 0; r < 8; ++r) {
      int   m = r + lg * 8;
      float v = acc[t][r] + bv;
      size_t idx;
      if (vmode) idx = ((size_t)(bl * 8 + h) * HD + hd) * NQ + (nq0 + m);
      else       idx = ((size_t)(bl * 8 + h) * NQ + (nq0 + m)) * HD + hd;
      out[idx] = (_Float16)v;
    }
  }
}

// ---------------------------------------------------------------------------
// Kernel 2: flash attention per (bl,h). grid = 768 blocks x 256 threads.
// K (24 KB) and V^T (24 KB) are staged once per block into LDS (the WGP's
// 320 KB LDS is the intended home for this working set on CDNA5); all WMMA
// B-fragments then come from ds_load_b128 instead of re-reading L2 ~24x.
// Per 32-key chunk: 2 QK^T WMMAs, online softmax (cross-lane max/sum over
// the 16-lane half-wave), LDS relayout of P into an A-fragment, 2 P*V WMMAs
// into the running O accumulator.
// ---------------------------------------------------------------------------
__global__ void __launch_bounds__(256)
attn_kernel(const _Float16* __restrict__ qw,  // [blh][384][32]
            const _Float16* __restrict__ kw,  // [blh][384][32]
            const _Float16* __restrict__ vt,  // [blh][32][384]
            _Float16* __restrict__ ow) {      // [bl*384][256]
  int blh  = blockIdx.x;
  int bl   = blh >> 3;
  int h    = blh & 7;
  int wid  = threadIdx.x >> 5;
  int lane = threadIdx.x & 31;
  int lg   = lane >> 4;
  int ln   = lane & 15;

  __shared__ _Float16 sK[NQ][HD];      // 24 KB : K rows (contiguous hd)
  __shared__ _Float16 sV[HD][NQ];      // 24 KB : V^T rows (contiguous keys)
  __shared__ _Float16 pT[8][16][32];   //  8 KB : per-wave P relayout staging

  const _Float16* qb = qw + (size_t)blh * NQ * HD;
  const _Float16* kb = kw + (size_t)blh * NQ * HD;
  const _Float16* vb = vt + (size_t)blh * HD * NQ;

  // ---- cooperative staging: 48 KB in 16B chunks, coalesced ----
  {
    const h8* gk = (const h8*)kb;
    const h8* gv = (const h8*)vb;
    h8* skp = (h8*)&sK[0][0];
    h8* svp = (h8*)&sV[0][0];
    const int CH = NQ * HD / 8;                   // 1536 chunks each
#pragma unroll 2
    for (int c = threadIdx.x; c < CH; c += 256) { skp[c] = gk[c]; }
#pragma unroll 2
    for (int c = threadIdx.x; c < CH; c += 256) { svp[c] = gv[c]; }
  }
  __syncthreads();

  const float scale = 0.1767766952966369f;   // 1/sqrt(32)

  for (int qt = wid; qt < NQ / 16; qt += 8) {
    int qbase = qt * 16;
    const _Float16* qrow = qb + (size_t)(qbase + ln) * HD + lg * 8;
    h16 Aq = ld_h16(qrow, qrow + 16);

    v8f  O0 = {}, O1 = {};
    float mrun[8], lsum[8];
#pragma unroll
    for (int r = 0; r < 8; ++r) { mrun[r] = -1e30f; lsum[r] = 0.0f; }

    for (int j = 0; j < NQ; j += 32) {
      // ---- scores: two 16-key tiles; B columns = K rows from LDS ----
      const _Float16* kr0 = &sK[j + ln][0] + lg * 8;
      const _Float16* kr1 = &sK[j + 16 + ln][0] + lg * 8;
      v8f S0 = {}, S1 = {};
      S0 = wmma_f16(Aq, ld_h16(kr0, kr0 + 16), S0);
      S1 = wmma_f16(Aq, ld_h16(kr1, kr1 + 16), S1);

      // ---- online softmax update (row m = r + 8*lg, cols = lanes) ----
#pragma unroll
      for (int r = 0; r < 8; ++r) {
        float a = S0[r] * scale, b = S1[r] * scale;
        float mx = fmaxf(a, b);
#pragma unroll
        for (int off = 1; off < 16; off <<= 1)
          mx = fmaxf(mx, __shfl_xor(mx, off, 32));
        float newm = fmaxf(mrun[r], mx);
        float corr = __expf(mrun[r] - newm);
        mrun[r] = newm;
        float p0 = __expf(a - newm);
        float p1 = __expf(b - newm);
        float s  = p0 + p1;
#pragma unroll
        for (int off = 1; off < 16; off <<= 1)
          s += __shfl_xor(s, off, 32);
        lsum[r] = lsum[r] * corr + s;
        O0[r] *= corr;
        O1[r] *= corr;
        // stage P (f16) into LDS in C-layout coordinates
        int m = r + lg * 8;
        pT[wid][m][ln]      = (_Float16)p0;
        pT[wid][m][16 + ln] = (_Float16)p1;
      }

      // ---- reload P as an A-fragment (DS ops are in-order within a wave) ----
      const _Float16* prow = &pT[wid][ln][0] + lg * 8;
      h16 Pf = ld_h16(prow, prow + 16);

      // ---- O += P * V  (sV rows: contiguous keys -> contiguous B columns) ----
      const _Float16* vr0 = &sV[ln][0] + j + lg * 8;        // hd 0..15
      const _Float16* vr1 = &sV[16 + ln][0] + j + lg * 8;   // hd 16..31
      O0 = wmma_f16(Pf, ld_h16(vr0, vr0 + 16), O0);
      O1 = wmma_f16(Pf, ld_h16(vr1, vr1 + 16), O1);
    }

    // ---- normalize and store o in [bl*384 + q][h*32 + hd] (f16) ----
#pragma unroll
    for (int r = 0; r < 8; ++r) {
      float inv = 1.0f / lsum[r];
      int   m   = r + lg * 8;
      size_t row = (size_t)(bl * NQ + qbase + m) * ATT + h * HD;
      ow[row + ln]      = (_Float16)(O0[r] * inv);
      ow[row + 16 + ln] = (_Float16)(O1[r] * inv);
    }
  }
}

// ---------------------------------------------------------------------------
// Kernel 3: output projection  out = o[36864,256] * Wd[256,128] + bd  (f32)
// One wave -> 16 rows x 32 cols, K = 256 in 8 WMMA steps.
// grid = 2304 blocks x 128 threads.
// ---------------------------------------------------------------------------
__global__ void outproj_kernel(const _Float16* __restrict__ O,    // [36864][256]
                               const _Float16* __restrict__ WdT,  // [128][256]
                               const float* __restrict__ bias,    // [128]
                               float* __restrict__ out) {         // [36864][128]
  int w     = blockIdx.x * 4 + (threadIdx.x >> 5);
  int lane  = threadIdx.x & 31;
  int lg    = lane >> 4;
  int ln    = lane & 15;
  int mtile = w >> 2;            // 0..2303
  int nbase = (w & 3) * 32;      // 0,32,64,96
  int rbase = mtile * 16;

  const _Float16* orow = O + (size_t)(rbase + ln) * ATT;
  v8f acc0 = {}, acc1 = {};

  for (int k = 0; k < ATT; k += 32) {
    const _Float16* p0 = orow + k + lg * 8;
    h16 A = ld_h16(p0, p0 + 16);
    const _Float16* w0 = WdT + (size_t)(nbase + ln) * ATT + k + lg * 8;
    const _Float16* w1 = WdT + (size_t)(nbase + 16 + ln) * ATT + k + lg * 8;
    acc0 = wmma_f16(A, ld_h16(w0, w0 + 16), acc0);
    acc1 = wmma_f16(A, ld_h16(w1, w1 + 16), acc1);
  }

  float b0 = bias[nbase + ln];
  float b1 = bias[nbase + 16 + ln];
#pragma unroll
  for (int r = 0; r < 8; ++r) {
    int m = r + lg * 8;
    float* po = out + (size_t)(rbase + m) * DMODEL;
    po[nbase + ln]      = acc0[r] + b0;
    po[nbase + 16 + ln] = acc1[r] + b1;
  }
}

// ---------------------------------------------------------------------------
// Host-side launcher
// ---------------------------------------------------------------------------
extern "C" void kernel_launch(void* const* d_in, const int* in_sizes, int n_in,
                              void* d_out, int out_size, void* d_ws, size_t ws_size,
                              hipStream_t stream) {
  (void)in_sizes; (void)n_in; (void)out_size; (void)ws_size;

  const float* query = (const float*)d_in[0];
  const float* key   = (const float*)d_in[1];
  const float* value = (const float*)d_in[2];
  const float* Wq    = (const float*)d_in[3];
  const float* bq    = (const float*)d_in[4];
  const float* Wk    = (const float*)d_in[5];
  const float* bk    = (const float*)d_in[6];
  const float* Wv    = (const float*)d_in[7];
  const float* bv    = (const float*)d_in[8];
  const float* Wd    = (const float*)d_in[9];
  const float* bd    = (const float*)d_in[10];
  float* out = (float*)d_out;

  // workspace layout (bytes)
  char* ws = (char*)d_ws;
  _Float16* WqT = (_Float16*)(ws + 0);        // 64 KB
  _Float16* WkT = (_Float16*)(ws + 65536);    // 64 KB
  _Float16* WvT = (_Float16*)(ws + 131072);   // 64 KB
  _Float16* WdT = (_Float16*)(ws + 196608);   // 64 KB
  const size_t QKV_BYTES = (size_t)BLH * NQ * HD * sizeof(_Float16);  // 18.87 MB
  _Float16* qws = (_Float16*)(ws + 262144);
  _Float16* kws = (_Float16*)((char*)qws + QKV_BYTES);
  _Float16* vws = (_Float16*)((char*)kws + QKV_BYTES);
  _Float16* ows = (_Float16*)((char*)vws + QKV_BYTES);
  // total ws use: 256 KB + 4 * 18.87 MB ~= 75.8 MB

  // 0) weight prep
  prep_weights_kernel<<<512, 256, 0, stream>>>(Wq, Wk, Wv, Wd, WqT, WkT, WvT, WdT);

  // 1) projections (q/k row-major per head; v transposed per head)
  proj_kernel<<<2304, 128, 0, stream>>>(query, WqT, bq, qws, 0);
  proj_kernel<<<2304, 128, 0, stream>>>(key,   WkT, bk, kws, 0);
  proj_kernel<<<2304, 128, 0, stream>>>(value, WvT, bv, vws, 1);

  // 2) flash attention over 768 (bl, h) problems
  attn_kernel<<<BLH, 256, 0, stream>>>(qws, kws, vws, ows);

  // 3) output projection + bias -> f32
  outproj_kernel<<<2304, 128, 0, stream>>>(ows, WdT, bd, out);
}
